// FusedCirculantBlock_1589137900081
// MI455X (gfx1250) — matrix operand order repaired and checked by reading, not compiled, not run on hardware
//
#include <hip/hip_runtime.h>

typedef __attribute__((ext_vector_type(16))) __bf16 v16bf;
typedef __attribute__((ext_vector_type(8)))  __bf16 v8bf;
typedef __attribute__((ext_vector_type(4)))  __bf16 v4bf;
typedef __attribute__((ext_vector_type(8)))  float  v8f;

#define DIM_M   8192      // B*S tokens
#define DIM_N   4096      // F_OUT
#define DIM_K   4096      // F_IN
#define QIN     8         // F_IN / 512
#define BM      128
#define BN      128
#define KS      32        // K per LDS stage (one bf16 WMMA k-step)
#define PITCH   40        // bf16 units: 80B rows -> 16B aligned, 20-dword bank stride

// y = GELU( (x .* sign) @ W_circ^T + bias )
// Split-bf16 GEMM: x=xh+xl, w=wh+wl (bf16), acc_f32 += xh*wh + xl*wh + xh*wl.
__global__ __launch_bounds__(256)
void circulant_gemm_bf16x3(const float* __restrict__ x,
                           const float* __restrict__ w,
                           const float* __restrict__ bias,
                           const int*   __restrict__ sbits,
                           float* __restrict__ out)
{
    __shared__ __bf16 Ah[BM][PITCH];
    __shared__ __bf16 Al[BM][PITCH];
    __shared__ __bf16 Bh[BN][PITCH];
    __shared__ __bf16 Bl[BN][PITCH];

    const int tid   = threadIdx.x;
    const int lane  = tid & 31;
    const int wv    = tid >> 5;
    const int waveM = wv & 1;        // 2 waves along M (64 rows each)
    const int waveN = wv >> 1;       // 4 waves along N (32 cols each)
    const int l16   = lane & 15;
    const int lhalf = lane >> 4;

    const int n0 = blockIdx.x * BN;
    const int m0 = blockIdx.y * BM;

    const int oblk = n0 >> 9;        // constant per tile (128 | 512)
    const int c0   = n0 & 511;

    v8f acc[4][2];
    const v8f vzero = {};
#pragma unroll
    for (int ms = 0; ms < 4; ++ms)
#pragma unroll
        for (int ns = 0; ns < 2; ++ns)
            acc[ms][ns] = vzero;

    for (int k0 = 0; k0 < DIM_K; k0 += KS) {
        const int iblk = k0 >> 9;
        const int mk0  = k0 & 511;
        const float* __restrict__ wblk = w + (((size_t)oblk * QIN + iblk) << 9);

        // ---- stage A: 128x32 activations, sign folded, split to bf16 hi/lo
#pragma unroll
        for (int j = 0; j < 4; ++j) {
            const int e   = tid + 256 * j;   // 0..1023 float4 slots
            const int row = e >> 3;
            const int q   = e & 7;
            const float4 xv = *(const float4*)(x + (size_t)(m0 + row) * DIM_K + k0 + q * 4);
            const int4   sv = *(const int4*)(sbits + k0 + q * 4);
            float f0 = xv.x * (float)(1 - 2 * sv.x);
            float f1 = xv.y * (float)(1 - 2 * sv.y);
            float f2 = xv.z * (float)(1 - 2 * sv.z);
            float f3 = xv.w * (float)(1 - 2 * sv.w);
            v4bf h, l;
            h[0] = (__bf16)f0; l[0] = (__bf16)(f0 - (float)h[0]);
            h[1] = (__bf16)f1; l[1] = (__bf16)(f1 - (float)h[1]);
            h[2] = (__bf16)f2; l[2] = (__bf16)(f2 - (float)h[2]);
            h[3] = (__bf16)f3; l[3] = (__bf16)(f3 - (float)h[3]);
            *(v4bf*)&Ah[row][q * 4] = h;
            *(v4bf*)&Al[row][q * 4] = l;
        }

        // ---- stage B: 32x128 circulant slab from the cached 512-float row, split hi/lo
        //      B[k,n] = wblk[(n%512 - k%512) mod 512]
#pragma unroll
        for (int j = 0; j < 4; ++j) {
            const int e  = tid + 256 * j;    // 0..1023 k-quads
            const int nn = e >> 3;
            const int kb = (e & 7) * 4;
            v4bf h, l;
#pragma unroll
            for (int t = 0; t < 4; ++t) {
                const int idx = (c0 + nn - mk0 - (kb + t) + 1024) & 511;
                const float f = wblk[idx];
                const __bf16 hh = (__bf16)f;
                h[t] = hh;
                l[t] = (__bf16)(f - (float)hh);
            }
            *(v4bf*)&Bh[nn][kb] = h;
            *(v4bf*)&Bl[nn][kb] = l;
        }

        // prefetch next activation k-slice (global_prefetch_b8)
        if (k0 + KS < DIM_K) {
            const int prow = tid >> 1;
            __builtin_prefetch(x + (size_t)(m0 + prow) * DIM_K + k0 + KS + (tid & 1) * 16, 0, 0);
        }

        __syncthreads();

        // ---- one 16x16x32 bf16 k-step, 3-term split accumulate
        v16bf ah[4], al[4], bh[2], bl[2];
#pragma unroll
        for (int ms = 0; ms < 4; ++ms) {
            const int r = waveM * 64 + ms * 16 + l16;
            // A VGPR layout: lanes 0-15 K {0..7,16..23}; lanes 16-31 K {8..15,24..31}
            v8bf h0 = *(const v8bf*)&Ah[r][8 * lhalf];
            v8bf h1 = *(const v8bf*)&Ah[r][16 + 8 * lhalf];
            ah[ms] = __builtin_shufflevector(h0, h1, 0,1,2,3,4,5,6,7,8,9,10,11,12,13,14,15);
            v8bf l0 = *(const v8bf*)&Al[r][8 * lhalf];
            v8bf l1 = *(const v8bf*)&Al[r][16 + 8 * lhalf];
            al[ms] = __builtin_shufflevector(l0, l1, 0,1,2,3,4,5,6,7,8,9,10,11,12,13,14,15);
        }
#pragma unroll
        for (int ns = 0; ns < 2; ++ns) {
            const int c = waveN * 32 + ns * 16 + l16;
            // B VGPR layout: lanes 0-15 K {0..7,16..23}; lanes 16-31 K {8..15,24..31}
            v8bf h0 = *(const v8bf*)&Bh[c][8 * lhalf];
            v8bf h1 = *(const v8bf*)&Bh[c][16 + 8 * lhalf];
            bh[ns] = __builtin_shufflevector(h0, h1, 0,1,2,3,4,5,6,7,8,9,10,11,12,13,14,15);
            v8bf l0 = *(const v8bf*)&Bl[c][8 * lhalf];
            v8bf l1 = *(const v8bf*)&Bl[c][16 + 8 * lhalf];
            bl[ns] = __builtin_shufflevector(l0, l1, 0,1,2,3,4,5,6,7,8,9,10,11,12,13,14,15);
        }
#pragma unroll
        for (int ms = 0; ms < 4; ++ms)
#pragma unroll
            for (int ns = 0; ns < 2; ++ns) {
                acc[ms][ns] = __builtin_amdgcn_wmma_f32_16x16x32_bf16(
                    false, ah[ms], false, bh[ns], (short)0, acc[ms][ns], false, false);
                acc[ms][ns] = __builtin_amdgcn_wmma_f32_16x16x32_bf16(
                    false, al[ms], false, bh[ns], (short)0, acc[ms][ns], false, false);
                acc[ms][ns] = __builtin_amdgcn_wmma_f32_16x16x32_bf16(
                    false, ah[ms], false, bl[ns], (short)0, acc[ms][ns], false, false);
            }
        __syncthreads();
    }

    // ---- epilogue: bias + exact erf-GELU, f32 stores
#pragma unroll
    for (int ms = 0; ms < 4; ++ms) {
#pragma unroll
        for (int ns = 0; ns < 2; ++ns) {
            const int gcol = n0 + waveN * 32 + ns * 16 + l16;
            const float bv = bias[gcol];
#pragma unroll
            for (int v = 0; v < 8; ++v) {
                const int grow = m0 + waveM * 64 + ms * 16 + v + 8 * lhalf;
                const float y = acc[ms][ns][v] + bv;
                const float g = 0.5f * y * (1.0f + erff(y * 0.7071067811865475f));
                out[(size_t)grow * DIM_N + gcol] = g;
            }
        }
    }
}

extern "C" void kernel_launch(void* const* d_in, const int* in_sizes, int n_in,
                              void* d_out, int out_size, void* d_ws, size_t ws_size,
                              hipStream_t stream) {
    const float* x     = (const float*)d_in[0];   // [2,4096,4096] f32
    const float* w     = (const float*)d_in[1];   // [8,8,512]     f32
    const float* bias  = (const float*)d_in[2];   // [4096]        f32
    const int*   sbits = (const int*)d_in[3];     // [4096]        i32 {0,1}
    float* out = (float*)d_out;                   // [2,4096,4096] f32

    dim3 grid(DIM_N / BN, DIM_M / BM);            // 32 x 64 tiles
    circulant_gemm_bf16x3<<<grid, dim3(256), 0, stream>>>(x, w, bias, sbits, out);
    (void)in_sizes; (void)n_in; (void)out_size; (void)d_ws; (void)ws_size;
}
